// PairwiseCost_18451179503711
// MI455X (gfx1250) — matrix operand order repaired and checked by reading, not compiled, not run on hardware
//
#include <hip/hip_runtime.h>

// ---------------------------------------------------------------------------
// Pairwise squared-Euclidean cost, B=16, N=2048, D=128 (fp32 in/out).
// Strategy: bf16x3 split-precision GEMM on v_wmma_f32_16x16x32_bf16,
// exact fp32 norms folded in the epilogue, NT stores for the 256MB output.
// Split uses truncation (not RNE): residual of a truncated-bf16 is exact in
// fp32, so hi/lo capture x to ~2^-17 relative at 3 VALU/float (perm/and/sub).
// ---------------------------------------------------------------------------

#define N_DIM 2048
#define D_DIM 128
#define TILE  128
#define LDS_STRIDE 136   // bf16 elements per row (+8 pad -> conflict-free b128)

typedef __attribute__((ext_vector_type(16))) __bf16        v16bf;
typedef __attribute__((ext_vector_type(8)))  float         v8f;
typedef __attribute__((ext_vector_type(4)))  float         f4v;
typedef __attribute__((ext_vector_type(4)))  unsigned int  u4v;
typedef __attribute__((ext_vector_type(2)))  unsigned int  u2v;

union Frag { u4v u[2]; v16bf v; };   // 32 bytes = one WMMA A/B fragment per lane

// Pack the high halfwords of (a, b) as {hi16(a), hi16(b)} -> one v_perm_b32.
// Result bits[15:0] = b[31:16], bits[31:16] = a[31:16].
__device__ __forceinline__ unsigned int pack_hi16(unsigned int a, unsigned int b) {
#if defined(__has_builtin) && __has_builtin(__builtin_amdgcn_perm)
    return __builtin_amdgcn_perm(a, b, 0x07060302u);
#else
    return (a & 0xFFFF0000u) | (b >> 16);
#endif
}

// Convert 4 floats -> packed bf16 hi pair-dwords + bf16 lo pair-dwords in LDS.
// hi = bf16-truncate(x); lo = bf16-truncate(x - hi). 12 VALU per float4.
__device__ __forceinline__ void cvt_store(unsigned short* __restrict__ hiArr,
                                          unsigned short* __restrict__ loArr,
                                          int row, int col, f4v val) {
    unsigned int u0 = __float_as_uint(val.x);
    unsigned int u1 = __float_as_uint(val.y);
    unsigned int u2 = __float_as_uint(val.z);
    unsigned int u3 = __float_as_uint(val.w);

    u2v hp;
    hp.x = pack_hi16(u1, u0);
    hp.y = pack_hi16(u3, u2);

    float r0 = val.x - __uint_as_float(u0 & 0xFFFF0000u);   // exact residual
    float r1 = val.y - __uint_as_float(u1 & 0xFFFF0000u);
    float r2 = val.z - __uint_as_float(u2 & 0xFFFF0000u);
    float r3 = val.w - __uint_as_float(u3 & 0xFFFF0000u);

    u2v lp;
    lp.x = pack_hi16(__float_as_uint(r1), __float_as_uint(r0));
    lp.y = pack_hi16(__float_as_uint(r3), __float_as_uint(r2));

    *(u2v*)(hiArr + row * LDS_STRIDE + col) = hp;
    *(u2v*)(loArr + row * LDS_STRIDE + col) = lp;
}

__global__ void __launch_bounds__(256)
pairwise_cost_wmma(const float* __restrict__ x,
                   const float* __restrict__ y,
                   float* __restrict__ out) {
    const int tid   = threadIdx.x;
    const int batch = blockIdx.z;
    const int iBase = blockIdx.x * TILE;   // output row tile
    const int jBase = blockIdx.y * TILE;   // output col tile

    const float* xB = x + ((size_t)batch * N_DIM + iBase) * D_DIM;
    const float* yB = y + ((size_t)batch * N_DIM + jBase) * D_DIM;

    extern __shared__ char smem[];
    unsigned short* sXhi = (unsigned short*)smem;            // [128][136] bf16
    unsigned short* sXlo = sXhi + TILE * LDS_STRIDE;
    unsigned short* sYhi = sXlo + TILE * LDS_STRIDE;
    unsigned short* sYlo = sYhi + TILE * LDS_STRIDE;
    float*          sXn  = (float*)(sYlo + TILE * LDS_STRIDE);  // [128] row norms
    float*          sYn  = sXn + TILE;                          // [128] col norms

    // ---- Load 128x128 X and Y tiles, split fp32 -> bf16 hi/lo into LDS ----
    // Fully coalesced: each iteration the workgroup reads 2 contiguous 4KB rows.
    #pragma unroll
    for (int it = 0; it < 16; ++it) {
        int f4  = it * 256 + tid;          // 0..4095 float4 slots of the tile
        int row = f4 >> 5;                 // / (128/4)
        int col = (f4 & 31) << 2;
        f4v vx = *(const f4v*)(xB + (size_t)row * D_DIM + col);
        f4v vy = *(const f4v*)(yB + (size_t)row * D_DIM + col);
        cvt_store(sXhi, sXlo, row, col, vx);
        cvt_store(sYhi, sYlo, row, col, vy);
    }

    // ---- Exact fp32 squared norms (re-read from cache-hot global) ----
    {
        int r = tid & 127;
        const float* src = (tid < 128) ? (xB + (size_t)r * D_DIM)
                                       : (yB + (size_t)r * D_DIM);
        float s = 0.0f;
        #pragma unroll
        for (int d = 0; d < D_DIM; d += 4) {
            f4v v = *(const f4v*)(src + d);
            s += v.x * v.x + v.y * v.y + v.z * v.z + v.w * v.w;
        }
        if (tid < 128) sXn[r] = s; else sYn[r] = s;
    }
    __syncthreads();

    // ---- Wave tiling: 8 waves, 2(M) x 4(N); wave tile = 64x32 ----
    const int lane = tid & 31;
    const int wave = tid >> 5;
    const int wm   = (wave & 1) * 64;      // wave row offset in tile
    const int wn   = (wave >> 1) * 32;     // wave col offset in tile
    const int lr   = lane & 15;            // row (A) / col (B) within 16-tile
    const int lh   = lane >> 4;            // lane-half select

    v8f acc[4][2];
    #pragma unroll
    for (int mt = 0; mt < 4; ++mt)
        #pragma unroll
        for (int nt = 0; nt < 2; ++nt)
            #pragma unroll
            for (int e = 0; e < 8; ++e) acc[mt][nt][e] = 0.0f;

    // ---- K loop: D=128 in 4 steps of K=32; bf16x3 per step ----
    #pragma unroll
    for (int kt = 0; kt < 4; ++kt) {
        const int kA = kt * 32 + lh * 8;    // A frag: K [kA..kA+7], [kA+16..kA+23]
        const int kB = kt * 32 + lh * 8;    // B frag: K [kB..kB+7], [kB+16..kB+23]

        Frag aHi[4], aLo[4];
        #pragma unroll
        for (int mt = 0; mt < 4; ++mt) {
            const int row = wm + mt * 16 + lr;
            const unsigned short* ph = sXhi + row * LDS_STRIDE;
            const unsigned short* pl = sXlo + row * LDS_STRIDE;
            aHi[mt].u[0] = *(const u4v*)(ph + kA);
            aHi[mt].u[1] = *(const u4v*)(ph + kA + 16);
            aLo[mt].u[0] = *(const u4v*)(pl + kA);
            aLo[mt].u[1] = *(const u4v*)(pl + kA + 16);
        }
        Frag bHi[2], bLo[2];
        #pragma unroll
        for (int nt = 0; nt < 2; ++nt) {
            const int colr = wn + nt * 16 + lr;   // Y row = output column
            const unsigned short* ph = sYhi + colr * LDS_STRIDE;
            const unsigned short* pl = sYlo + colr * LDS_STRIDE;
            bHi[nt].u[0] = *(const u4v*)(ph + kB);
            bHi[nt].u[1] = *(const u4v*)(ph + kB + 16);
            bLo[nt].u[0] = *(const u4v*)(pl + kB);
            bLo[nt].u[1] = *(const u4v*)(pl + kB + 16);
        }

        #pragma unroll
        for (int mt = 0; mt < 4; ++mt) {
            #pragma unroll
            for (int nt = 0; nt < 2; ++nt) {
                acc[mt][nt] = __builtin_amdgcn_wmma_f32_16x16x32_bf16(
                    false, aHi[mt].v, false, bHi[nt].v, (short)0, acc[mt][nt], false, false);
                acc[mt][nt] = __builtin_amdgcn_wmma_f32_16x16x32_bf16(
                    false, aHi[mt].v, false, bLo[nt].v, (short)0, acc[mt][nt], false, false);
                acc[mt][nt] = __builtin_amdgcn_wmma_f32_16x16x32_bf16(
                    false, aLo[mt].v, false, bHi[nt].v, (short)0, acc[mt][nt], false, false);
            }
        }
    }

    // ---- Epilogue: cost = max(0, ||x||^2 + ||y||^2 - 2*dot), NT streamed ----
    float* outB = out + (size_t)batch * N_DIM * N_DIM;
    #pragma unroll
    for (int mt = 0; mt < 4; ++mt) {
        #pragma unroll
        for (int nt = 0; nt < 2; ++nt) {
            const int jLocal = wn + nt * 16 + lr;   // C/D layout: N = lane%16
            const float yn = sYn[jLocal];
            const size_t j = (size_t)(jBase + jLocal);
            #pragma unroll
            for (int e = 0; e < 8; ++e) {           // M = e + 8*(lane/16)
                const int mLocal = wm + mt * 16 + lh * 8 + e;
                const float c =
                    fmaxf(sXn[mLocal] + yn - 2.0f * acc[mt][nt][e], 0.0f);
                __builtin_nontemporal_store(
                    c, outB + (size_t)(iBase + mLocal) * N_DIM + j);
            }
        }
    }
}

extern "C" void kernel_launch(void* const* d_in, const int* in_sizes, int n_in,
                              void* d_out, int out_size, void* d_ws, size_t ws_size,
                              hipStream_t stream) {
    const float* x = (const float*)d_in[0];
    const float* y = (const float*)d_in[1];
    float* out = (float*)d_out;

    const int B = in_sizes[0] / (N_DIM * D_DIM);   // = 16

    const size_t smemBytes =
        (size_t)4 * TILE * LDS_STRIDE * sizeof(unsigned short) +  // hi/lo tiles
        (size_t)2 * TILE * sizeof(float);                         // norms

    // Allow >64KB dynamic LDS (WGP supports up to 320KB per workgroup).
    // Unconditional: deterministic, not a stream op, graph-capture safe.
    hipFuncSetAttribute((const void*)pairwise_cost_wmma,
                        hipFuncAttributeMaxDynamicSharedMemorySize,
                        (int)smemBytes);

    dim3 grid(N_DIM / TILE, N_DIM / TILE, B);   // (16,16,16)
    pairwise_cost_wmma<<<grid, 256, smemBytes, stream>>>(x, y, out);
}
